// MainLoss_78786880078073
// MI455X (gfx1250) — matrix operand order repaired and checked by reading, not compiled, not run on hardware
//
#include <hip/hip_runtime.h>
#include <hip/hip_bf16.h>
#include <math.h>

// Problem constants (from reference)
constexpr int T = 2;
constexpr int N = 50000;
constexpr int D = 128;
constexpr int P = 10;
constexpr int K = 20;
constexpr float Q = 10.0f;
constexpr float EPS = 1e-8f;

typedef _Float16 v4h  __attribute__((ext_vector_type(4)));
typedef _Float16 v8h  __attribute__((ext_vector_type(8)));
typedef _Float16 v16h __attribute__((ext_vector_type(16)));
typedef float    v8f  __attribute__((ext_vector_type(8)));

__device__ __forceinline__ float log_sigmoid(float x) {
    // stable: min(x,0) - log1p(exp(-|x|)); |x| <= ~1 here (cosine sims)
    return fminf(x, 0.0f) - log1pf(__expf(-fabsf(x)));
}

// ---------------------------------------------------------------------------
// Kernel 1: L2-normalize each D=128 row, store as f16 staging table in ws.
// One wave per node; each lane owns 4 floats (32 lanes * 4 = 128).
// ---------------------------------------------------------------------------
__global__ void normalize_f16_kernel(const float* __restrict__ emb,
                                     _Float16* __restrict__ e16) {
    const int lane = threadIdx.x & 31;
    const int node = blockIdx.x * (blockDim.x >> 5) + (threadIdx.x >> 5); // exact grid
    const float4 v = ((const float4*)(emb + (size_t)node * D))[lane];
    float ss = v.x * v.x + v.y * v.y + v.z * v.z + v.w * v.w;
    #pragma unroll
    for (int off = 16; off > 0; off >>= 1) ss += __shfl_xor(ss, off, 32);
    const float scale = 1.0f / fmaxf(sqrtf(ss), EPS);
    v4h o;
    o[0] = (_Float16)(v.x * scale);
    o[1] = (_Float16)(v.y * scale);
    o[2] = (_Float16)(v.z * scale);
    o[3] = (_Float16)(v.w * scale);
    ((v4h*)(e16 + (size_t)node * D))[lane] = o;
}

// ---------------------------------------------------------------------------
// Kernel 2: one wave per node. Scores of the 30 neighbors via WMMA:
//   A (16x32 f16, two M-tiles) = neighbor rows, B (32x16) = node emb
//   broadcast to all 16 columns, C accumulated f32 over 4 K-steps.
// C layout: VGPR r -> M=r (lanes 0-15) / M=r+8 (lanes 16-31); all columns equal.
// ---------------------------------------------------------------------------
__global__ void __launch_bounds__(256)
score_loss_kernel(const _Float16* __restrict__ e16,
                  const int* __restrict__ pos_idx,
                  const int* __restrict__ neg_idx,
                  float* __restrict__ partials) {
    const int lane = threadIdx.x & 31;
    const int w    = threadIdx.x >> 5;           // wave in block: 0..7
    const int g    = blockIdx.x * 8 + w;         // global node id, 0..T*N-1 exactly
    const int t    = g / N;
    const int n    = g - t * N;
    const int h    = lane >> 4;                  // K-half select (0/1)
    const int m    = lane & 15;                  // matrix row within tile

    const _Float16* __restrict__ eb   = e16 + (size_t)g * D;       // own row
    const _Float16* __restrict__ base = e16 + (size_t)t * N * D;   // timestamp slab

    // Neighbor row indices for the two 16-row tiles (rows 30,31 pad with self).
    const int* pp = pos_idx + (size_t)g * P;
    const int* nn = neg_idx + (size_t)g * K;
    const int r0 = m;                 // tile0 rows 0..15 : 10 pos + 6 neg
    const int r1 = 16 + m;            // tile1 rows 16..31: 14 neg + 2 pad
    const int idx0 = (r0 < P) ? pp[r0] : nn[r0 - P];
    const int idx1 = (r1 < P + K) ? nn[r1 - P] : n;
    const _Float16* a0p = base + (size_t)idx0 * D;
    const _Float16* a1p = base + (size_t)idx1 * D;

    v8f c0 = {};
    v8f c1 = {};
    #pragma unroll
    for (int kk = 0; kk < 4; ++kk) {
        const int kb = 32 * kk;
        // B fragment: K = kb+8h+{0..7} and kb+16+8h+{0..7} (16B-aligned loads)
        v16h b;
        {
            const v8h b0 = *(const v8h*)(eb + kb + 8 * h);
            const v8h b1 = *(const v8h*)(eb + kb + 16 + 8 * h);
            #pragma unroll
            for (int i = 0; i < 8; ++i) { b[i] = b0[i]; b[8 + i] = b1[i]; }
        }
        // A fragments: lane holds row m, K = kb+8h+{0..7} and kb+16+8h+{0..7}
        v16h a0, a1;
        {
            const v8h lo = *(const v8h*)(a0p + kb + 8 * h);
            const v8h hi = *(const v8h*)(a0p + kb + 16 + 8 * h);
            #pragma unroll
            for (int i = 0; i < 8; ++i) { a0[i] = lo[i]; a0[8 + i] = hi[i]; }
        }
        {
            const v8h lo = *(const v8h*)(a1p + kb + 8 * h);
            const v8h hi = *(const v8h*)(a1p + kb + 16 + 8 * h);
            #pragma unroll
            for (int i = 0; i < 8; ++i) { a1[i] = lo[i]; a1[8 + i] = hi[i]; }
        }
        c0 = __builtin_amdgcn_wmma_f32_16x16x32_f16(false, a0, false, b,
                                                    (short)0, c0, false, false);
        c1 = __builtin_amdgcn_wmma_f32_16x16x32_f16(false, a1, false, b,
                                                    (short)0, c1, false, false);
    }

    // Per-lane log-sigmoid partials over the rows this half-wave owns.
    float pos = 0.0f, neg = 0.0f;
    if (h == 0) {
        // tile0 rows 0..7 -> positives; tile1 rows 16..23 -> negatives
        #pragma unroll
        for (int r = 0; r < 8; ++r) pos += log_sigmoid(c0[r]);
        #pragma unroll
        for (int r = 0; r < 8; ++r) neg += log_sigmoid(-c1[r]);
    } else {
        // tile0 rows 8..15: rows 8,9 pos; 10..15 neg. tile1 rows 24..29 neg; 30,31 pad.
        pos += log_sigmoid(c0[0]) + log_sigmoid(c0[1]);
        #pragma unroll
        for (int r = 2; r < 8; ++r) neg += log_sigmoid(-c0[r]);
        #pragma unroll
        for (int r = 0; r < 6; ++r) neg += log_sigmoid(-c1[r]);
    }
    // Combine the two half-wave partials (columns were identical within halves).
    pos += __shfl_xor(pos, 16, 32);
    neg += __shfl_xor(neg, 16, 32);

    const float node_loss = -(pos * (1.0f / P)) - Q * (neg * (1.0f / K));

    __shared__ float sblk[8];
    if (lane == 0) sblk[w] = node_loss;
    __syncthreads();
    if (threadIdx.x == 0) {
        float s = 0.0f;
        #pragma unroll
        for (int i = 0; i < 8; ++i) s += sblk[i];
        partials[blockIdx.x] = s;  // fixed order -> deterministic
    }
}

// ---------------------------------------------------------------------------
// Kernel 3: deterministic fixed-order reduction of block partials -> scalar.
// ---------------------------------------------------------------------------
__global__ void reduce_kernel(const float* __restrict__ partials, int nblocks,
                              float* __restrict__ out) {
    __shared__ float s[256];
    float acc = 0.0f;
    for (int i = threadIdx.x; i < nblocks; i += 256) acc += partials[i];
    s[threadIdx.x] = acc;
    __syncthreads();
    for (int off = 128; off > 0; off >>= 1) {
        if (threadIdx.x < off) s[threadIdx.x] += s[threadIdx.x + off];
        __syncthreads();
    }
    if (threadIdx.x == 0) out[0] = s[0] * (1.0f / ((float)T * (float)N));
}

extern "C" void kernel_launch(void* const* d_in, const int* in_sizes, int n_in,
                              void* d_out, int out_size, void* d_ws, size_t ws_size,
                              hipStream_t stream) {
    const float* emb     = (const float*)d_in[0];
    const int*   pos_idx = (const int*)d_in[1];
    const int*   neg_idx = (const int*)d_in[2];
    float*       out     = (float*)d_out;

    _Float16* e16 = (_Float16*)d_ws;                       // 25.6 MB f16 table
    const size_t tbl_bytes = (size_t)T * N * D * sizeof(_Float16);
    float* partials = (float*)((char*)d_ws + ((tbl_bytes + 255) & ~(size_t)255));

    const int nodes = T * N;           // 100000
    const int nblk  = nodes / 8;       // 12500 blocks of 8 waves (exact)

    normalize_f16_kernel<<<nblk, 256, 0, stream>>>(emb, e16);
    score_loss_kernel<<<nblk, 256, 0, stream>>>(e16, pos_idx, neg_idx, partials);
    reduce_kernel<<<1, 256, 0, stream>>>(partials, nblk, out);
}
